// DynamicDeepHitMambaCompeting_62182536511818
// MI455X (gfx1250) — compile-verified, matrix operand order checked by reading, not run on hardware
//
#include <hip/hip_runtime.h>

typedef __attribute__((ext_vector_type(16))) __bf16 v16bf;
typedef __attribute__((ext_vector_type(8)))  float  v8f;

constexpr int Bn = 64, Ln = 512, Fn = 41, Hn = 128, NLn = 2;
constexpr int DSn = 16, DCn = 4, DIn = 256, DTRn = 8, En = 3, Tn = 100, FCn = 160;
constexpr float SCALEc = 0.08838834764831845f;   // 1/sqrt(128)

// ---------------------------------------------------------------- utilities

__global__ void f32_to_bf16_kernel(const float* __restrict__ src,
                                   __bf16* __restrict__ dst, int n) {
  int i = blockIdx.x * blockDim.x + threadIdx.x;
  if (i < n) dst[i] = (__bf16)src[i];
}

// ------------------------------------------------------- input projections

// h[b,l,:] = x[b,l,:] @ ip_W^T + ip_b      (grid = B*L, block = H)
__global__ void input_proj_kernel(const float* __restrict__ x,
                                  const float* __restrict__ W,
                                  const float* __restrict__ bias,
                                  float* __restrict__ h) {
  int row = blockIdx.x;
  int t = threadIdx.x;              // 0..127
  __shared__ float xs[Fn];
  if (t < Fn) xs[t] = x[(size_t)row * Fn + t];
  __syncthreads();
  float acc = bias[t];
  #pragma unroll
  for (int f = 0; f < Fn; ++f) acc += xs[f] * W[t * Fn + f];
  h[(size_t)row * Hn + t] = acc;
}

// last[b,:] = x[b,L-1,:] @ ipr_W^T + ipr_b   (grid = B, block = H)
__global__ void last_proj_kernel(const float* __restrict__ x,
                                 const float* __restrict__ W,
                                 const float* __restrict__ bias,
                                 float* __restrict__ last) {
  int b = blockIdx.x;
  int t = threadIdx.x;
  __shared__ float xs[Fn];
  size_t row = (size_t)b * Ln + (Ln - 1);
  if (t < Fn) xs[t] = x[row * Fn + t];
  __syncthreads();
  float acc = bias[t];
  #pragma unroll
  for (int f = 0; f < Fn; ++f) acc += xs[f] * W[t * Fn + f];
  last[(size_t)b * Hn + t] = acc;
}

// ---------------------------------------------------------------- layernorm
// grid = B*L, block = H.  Writes f32 and/or bf16 (null disables).
__global__ void ln_kernel(const float* __restrict__ in,
                          const float* __restrict__ g,
                          const float* __restrict__ bta,
                          float* __restrict__ outf,
                          __bf16* __restrict__ outb) {
  int row = blockIdx.x;
  int t = threadIdx.x;              // 128
  __shared__ float red[Hn];
  float v = in[(size_t)row * Hn + t];
  red[t] = v; __syncthreads();
  for (int s = 64; s > 0; s >>= 1) { if (t < s) red[t] += red[t + s]; __syncthreads(); }
  float mu = red[0] * (1.0f / Hn);
  __syncthreads();
  float d = v - mu;
  red[t] = d * d; __syncthreads();
  for (int s = 64; s > 0; s >>= 1) { if (t < s) red[t] += red[t + s]; __syncthreads(); }
  float var = red[0] * (1.0f / Hn);
  float y = d * rsqrtf(var + 1e-5f) * g[t] + bta[t];
  if (outf) outf[(size_t)row * Hn + t] = y;
  if (outb) outb[(size_t)row * Hn + t] = (__bf16)y;
}

// ------------------------------------------------------------- WMMA GEMM
// C[M,N] = A[M,K] * W[N,K]^T   (A,W bf16 row-major, f32 accumulate)
// One 16x16 C-tile per wave, 8 waves per block.
// A-operand layout (16x32 bf16): lane m / m+16 holds row m, two contiguous
// 8-element K-chunks at 8*half and 8*half+16.  B operand mirrored with
// lane indexing N — matches a row-major W[N][K].
template<bool ACCUM, bool STORE_BF16>
__global__ __launch_bounds__(256) void gemm_wmma_kernel(
    const __bf16* __restrict__ A, const __bf16* __restrict__ W,
    float* __restrict__ Cf, __bf16* __restrict__ Cb,
    int K, int N, int ntn, int total_tiles) {
  int wave = threadIdx.x >> 5;
  int tile = blockIdx.x * 8 + wave;
  if (tile >= total_tiles) return;
  int lane = threadIdx.x & 31;
  int half = lane >> 4;
  int idx  = lane & 15;
  int tm = (tile / ntn) << 4;
  int tn = (tile % ntn) << 4;
  const __bf16* Arow = A + (size_t)(tm + idx) * K;
  const __bf16* Wrow = W + (size_t)(tn + idx) * K;
  v8f acc = {};
  for (int k0 = 0; k0 < K; k0 += 32) {
    union { v16bf v; uint4 q[2]; } av, bv;
    av.q[0] = *(const uint4*)(Arow + k0 + 8 * half);
    av.q[1] = *(const uint4*)(Arow + k0 + 8 * half + 16);
    bv.q[0] = *(const uint4*)(Wrow + k0 + 8 * half);
    bv.q[1] = *(const uint4*)(Wrow + k0 + 8 * half + 16);
    acc = __builtin_amdgcn_wmma_f32_16x16x32_bf16(
        false, av.v, false, bv.v, (short)0, acc, false, false);
  }
  int mb = half * 8;
  #pragma unroll
  for (int r = 0; r < 8; ++r) {
    size_t o = (size_t)(tm + mb + r) * N + tn + idx;
    float v = acc[r];
    if (ACCUM) v += Cf[o];
    if (STORE_BF16) Cb[o] = (__bf16)v;
    else            Cf[o] = v;
  }
}

// ------------------------------------------------- depthwise causal conv+silu
// u[b,l,d] = silu( sum_j xz_u[b,l-3+j,d]*convW[d,j] + convb[d] )
// grid = B*L, block = DI
__global__ void conv_silu_kernel(const __bf16* __restrict__ xz,
                                 const float* __restrict__ convW,
                                 const float* __restrict__ convb,
                                 float* __restrict__ u) {
  int bl = blockIdx.x;
  int d  = threadIdx.x;             // 256
  int l  = bl % Ln;
  int base = bl - l;                // row index of (b, 0)
  const float* w = convW + d * DCn;
  float acc = convb[d];
  #pragma unroll
  for (int j = 0; j < DCn; ++j) {
    int lj = l - (DCn - 1) + j;
    if (lj >= 0)
      acc += (float)xz[(size_t)(base + lj) * (2 * DIn) + d] * w[j];
  }
  float s = acc / (1.0f + __expf(-acc));        // silu
  u[(size_t)bl * DIn + d] = s;
}

// xdb[b,l,r] = u[b,l,:] @ xp_W[r,:]     (grid = B*L, block = 64)
__global__ void xproj_kernel(const float* __restrict__ u,
                             const float* __restrict__ xpW,
                             float* __restrict__ xdb) {
  int bl = blockIdx.x;
  int t  = threadIdx.x;
  __shared__ float us[DIn];
  for (int i = t; i < DIn; i += 64) us[i] = u[(size_t)bl * DIn + i];
  __syncthreads();
  if (t < DTRn + 2 * DSn) {
    const float* w = xpW + t * DIn;
    float acc = 0.f;
    for (int d = 0; d < DIn; ++d) acc += us[d] * w[d];
    xdb[(size_t)bl * (DTRn + 2 * DSn) + t] = acc;
  }
}

// dt[b,l,d] = softplus( xdb[b,l,:8] @ dt_W[d,:] + dt_b[d] )  (grid=B*L, block=DI)
__global__ void dt_kernel(const float* __restrict__ xdb,
                          const float* __restrict__ dtW,
                          const float* __restrict__ dtb,
                          float* __restrict__ dt) {
  int bl = blockIdx.x;
  int d  = threadIdx.x;
  __shared__ float r[DTRn];
  if (d < DTRn) r[d] = xdb[(size_t)bl * (DTRn + 2 * DSn) + d];
  __syncthreads();
  float acc = dtb[d];
  #pragma unroll
  for (int j = 0; j < DTRn; ++j) acc += r[j] * dtW[d * DTRn + j];
  float sp = (acc > 20.f) ? acc : log1pf(__expf(acc));
  dt[(size_t)bl * DIn + d] = sp;
}

// ------------------------------------------------------------ selective scan
// State h[d,0:16] lives in VGPRs; y = (scan + u*D) * silu(z) -> bf16.
// grid = B, block = DI (one sequence per block, channel per thread).
__global__ __launch_bounds__(256) void scan_kernel(
    const float* __restrict__ u, const float* __restrict__ dt,
    const float* __restrict__ xdb, const __bf16* __restrict__ xz,
    const float* __restrict__ A_log, const float* __restrict__ Dp,
    __bf16* __restrict__ ybf) {
  int b = blockIdx.x;
  int d = threadIdx.x;
  float A[DSn];
  #pragma unroll
  for (int n = 0; n < DSn; ++n) A[n] = -__expf(A_log[d * DSn + n]);
  float Dd = Dp[d];
  float hs[DSn];
  #pragma unroll
  for (int n = 0; n < DSn; ++n) hs[n] = 0.f;
  __shared__ float Bs[DSn], Cs[DSn];
  for (int t = 0; t < Ln; ++t) {
    size_t bl = (size_t)b * Ln + t;
    size_t xo = bl * (DTRn + 2 * DSn);
    if (d < DSn)            Bs[d]        = xdb[xo + DTRn + d];
    else if (d < 2 * DSn)   Cs[d - DSn]  = xdb[xo + DTRn + DSn + (d - DSn)];
    __syncthreads();
    float ut  = u[bl * DIn + d];
    float dtt = dt[bl * DIn + d];
    float du  = dtt * ut;
    float y = 0.f;
    #pragma unroll
    for (int n = 0; n < DSn; ++n) {
      float dA = __expf(dtt * A[n]);
      hs[n] = dA * hs[n] + du * Bs[n];
      y += hs[n] * Cs[n];
    }
    float zv = (float)xz[bl * (2 * DIn) + DIn + d];
    float gate = zv / (1.0f + __expf(-zv));
    ybf[bl * DIn + d] = (__bf16)((y + ut * Dd) * gate);
    __syncthreads();
  }
}

// ------------------------------------------------------- attention (fused)
// kq[b,h] = sum_k (last@qW^T+qb)[k] * kW[k,h];  qkb[b] = query . kb
__global__ void attn_prep_kernel(const float* __restrict__ last,
                                 const float* __restrict__ qW,
                                 const float* __restrict__ qb,
                                 const float* __restrict__ kW,
                                 const float* __restrict__ kb,
                                 float* __restrict__ kq,
                                 float* __restrict__ qkb) {
  int b = blockIdx.x;
  int t = threadIdx.x;              // 128
  __shared__ float ls[Hn], qs[Hn];
  ls[t] = last[(size_t)b * Hn + t];
  __syncthreads();
  float acc = qb[t];
  for (int j = 0; j < Hn; ++j) acc += ls[j] * qW[t * Hn + j];
  qs[t] = acc;
  __syncthreads();
  float a2 = 0.f;
  for (int k = 0; k < Hn; ++k) a2 += qs[k] * kW[k * Hn + t];
  kq[(size_t)b * Hn + t] = a2;
  if (t == 0) {
    float s = 0.f;
    for (int k = 0; k < Hn; ++k) s += qs[k] * kb[k];
    qkb[b] = s;
  }
}

// scores -> masked softmax -> seq_repr[b,:] = sum_l w_l * hfin + last
// grid = B, block = 256
__global__ __launch_bounds__(256) void attn_pool_kernel(
    const float* __restrict__ hfin, const float* __restrict__ x,
    const float* __restrict__ kq, const float* __restrict__ qkb,
    const float* __restrict__ last, float* __restrict__ seq) {
  int b = blockIdx.x;
  int t = threadIdx.x;
  __shared__ float sc[Ln];
  __shared__ float kqs[Hn];
  __shared__ float red[256];
  if (t < Hn) kqs[t] = kq[(size_t)b * Hn + t];
  __syncthreads();
  float qk = qkb[b];
  for (int l = t; l < Ln; l += 256) {
    const float* hr = hfin + ((size_t)b * Ln + l) * Hn;
    float s = qk;
    for (int j = 0; j < Hn; ++j) s += hr[j] * kqs[j];
    s *= SCALEc;
    float m = x[((size_t)b * Ln + l) * Fn + (Fn - 1)];
    sc[l] = (m > 0.5f) ? s : -1e9f;
  }
  __syncthreads();
  float mx = -1e30f;
  for (int l = t; l < Ln; l += 256) mx = fmaxf(mx, sc[l]);
  red[t] = mx; __syncthreads();
  for (int s = 128; s > 0; s >>= 1) { if (t < s) red[t] = fmaxf(red[t], red[t + s]); __syncthreads(); }
  float mxv = red[0];
  __syncthreads();
  float sum = 0.f;
  for (int l = t; l < Ln; l += 256) { float e = __expf(sc[l] - mxv); sc[l] = e; sum += e; }
  red[t] = sum; __syncthreads();
  for (int s = 128; s > 0; s >>= 1) { if (t < s) red[t] += red[t + s]; __syncthreads(); }
  float inv = 1.0f / red[0];
  __syncthreads();
  if (t < Hn) {
    float acc = last[(size_t)b * Hn + t];
    for (int l = 0; l < Ln; ++l)
      acc += sc[l] * inv * hfin[((size_t)b * Ln + l) * Hn + t];
    seq[(size_t)b * Hn + t] = acc;
  }
}

// ----------------------------------------------------------------- head
// hid = gelu(seq @ h1W^T + h1b); logits = hid @ h2W^T + h2b   (grid=B,block=256)
__global__ __launch_bounds__(256) void head_kernel(
    const float* __restrict__ seq, const float* __restrict__ h1W,
    const float* __restrict__ h1b, const float* __restrict__ h2W,
    const float* __restrict__ h2b, float* __restrict__ out) {
  int b = blockIdx.x;
  int t = threadIdx.x;
  __shared__ float ss[Hn];
  __shared__ float hid[En * FCn];   // 480
  if (t < Hn) ss[t] = seq[(size_t)b * Hn + t];
  __syncthreads();
  for (int i = t; i < En * FCn; i += 256) {
    const float* w = h1W + (size_t)i * Hn;
    float acc = h1b[i];
    for (int j = 0; j < Hn; ++j) acc += ss[j] * w[j];
    hid[i] = 0.5f * acc * (1.0f + erff(acc * 0.70710678118654752f));
  }
  __syncthreads();
  for (int i = t; i < En * Tn; i += 256) {
    int e = i / Tn;
    const float* w  = h2W + (size_t)i * FCn;
    const float* hh = hid + e * FCn;
    float acc = h2b[i];
    for (int f = 0; f < FCn; ++f) acc += hh[f] * w[f];
    out[(size_t)b * En * Tn + i] = acc;
  }
}

// ================================================================= launcher

extern "C" void kernel_launch(void* const* d_in, const int* in_sizes, int n_in,
                              void* d_out, int out_size, void* d_ws, size_t ws_size,
                              hipStream_t stream) {
  const float* x      = (const float*)d_in[0];
  const float* ip_W   = (const float*)d_in[1];
  const float* ip_b   = (const float*)d_in[2];
  const float* ipr_W  = (const float*)d_in[3];
  const float* ipr_b  = (const float*)d_in[4];
  const float* ln_g   = (const float*)d_in[5];
  const float* ln_b   = (const float*)d_in[6];
  const float* in_W   = (const float*)d_in[7];
  const float* conv_W = (const float*)d_in[8];
  const float* conv_b = (const float*)d_in[9];
  const float* xp_W   = (const float*)d_in[10];
  const float* dt_W   = (const float*)d_in[11];
  const float* dt_b   = (const float*)d_in[12];
  const float* A_log  = (const float*)d_in[13];
  const float* Dp     = (const float*)d_in[14];
  const float* out_W  = (const float*)d_in[15];
  const float* on_g   = (const float*)d_in[16];
  const float* on_b   = (const float*)d_in[17];
  const float* q_W    = (const float*)d_in[18];
  const float* q_b    = (const float*)d_in[19];
  const float* k_W    = (const float*)d_in[20];
  const float* k_b    = (const float*)d_in[21];
  const float* h1_W   = (const float*)d_in[22];
  const float* h1_b   = (const float*)d_in[23];
  const float* h2_W   = (const float*)d_in[24];
  const float* h2_b   = (const float*)d_in[25];
  float* outp = (float*)d_out;

  const size_t BL = (size_t)Bn * Ln;

  char* ws = (char*)d_ws;
  size_t off = 0;
  auto alloc = [&](size_t bytes) -> void* {
    off = (off + 255) & ~(size_t)255;
    void* p = ws + off;
    off += bytes;
    return p;
  };

  float*  h     = (float*)alloc(BL * Hn * sizeof(float));          // 16 MB
  __bf16* a_bf  = (__bf16*)alloc(BL * DIn * sizeof(__bf16));       // 16 MB (h_ln / y)
  __bf16* xz_bf = (__bf16*)alloc(BL * 2 * DIn * sizeof(__bf16));   // 32 MB
  float*  u     = (float*)alloc(BL * DIn * sizeof(float));         // 32 MB (also hfin)
  float*  dt    = (float*)alloc(BL * DIn * sizeof(float));         // 32 MB
  float*  xdb   = (float*)alloc(BL * (DTRn + 2 * DSn) * sizeof(float)); // 5.2 MB
  __bf16* inWb  = (__bf16*)alloc((size_t)NLn * 2 * DIn * Hn * sizeof(__bf16));
  __bf16* outWb = (__bf16*)alloc((size_t)NLn * Hn * DIn * sizeof(__bf16));
  float*  last  = (float*)alloc((size_t)Bn * Hn * sizeof(float));
  float*  kq    = (float*)alloc((size_t)Bn * Hn * sizeof(float));
  float*  qkb   = (float*)alloc((size_t)Bn * sizeof(float));
  float*  seq   = (float*)alloc((size_t)Bn * Hn * sizeof(float));
  float*  hfin  = u;   // reuse after layers are done

  // weights -> bf16
  {
    int n1 = NLn * 2 * DIn * Hn;   // 131072
    int n2 = NLn * Hn * DIn;       // 65536
    f32_to_bf16_kernel<<<(n1 + 255) / 256, 256, 0, stream>>>(in_W, inWb, n1);
    f32_to_bf16_kernel<<<(n2 + 255) / 256, 256, 0, stream>>>(out_W, outWb, n2);
  }

  input_proj_kernel<<<(int)BL, Hn, 0, stream>>>(x, ip_W, ip_b, h);
  last_proj_kernel<<<Bn, Hn, 0, stream>>>(x, ipr_W, ipr_b, last);

  for (int i = 0; i < NLn; ++i) {
    // layernorm -> bf16
    ln_kernel<<<(int)BL, Hn, 0, stream>>>(h, ln_g + i * Hn, ln_b + i * Hn,
                                          nullptr, a_bf);
    // xz = h_ln @ in_W^T   (M=BL, K=128, N=512) -> bf16
    {
      int N = 2 * DIn, K = Hn;
      int ntn = N / 16, tiles = (int)(BL / 16) * ntn;
      gemm_wmma_kernel<false, true><<<(tiles + 7) / 8, 256, 0, stream>>>(
          a_bf, inWb + (size_t)i * 2 * DIn * Hn, nullptr, xz_bf, K, N, ntn, tiles);
    }
    conv_silu_kernel<<<(int)BL, DIn, 0, stream>>>(
        xz_bf, conv_W + i * DIn * DCn, conv_b + i * DIn, u);
    xproj_kernel<<<(int)BL, 64, 0, stream>>>(u, xp_W + i * (DTRn + 2 * DSn) * DIn, xdb);
    dt_kernel<<<(int)BL, DIn, 0, stream>>>(xdb, dt_W + i * DIn * DTRn,
                                           dt_b + i * DIn, dt);
    scan_kernel<<<Bn, DIn, 0, stream>>>(u, dt, xdb, xz_bf,
                                        A_log + i * DIn * DSn, Dp + i * DIn, a_bf);
    // h += y @ out_W^T   (M=BL, K=256, N=128), f32 accumulate into residual
    {
      int N = Hn, K = DIn;
      int ntn = N / 16, tiles = (int)(BL / 16) * ntn;
      gemm_wmma_kernel<true, false><<<(tiles + 7) / 8, 256, 0, stream>>>(
          a_bf, outWb + (size_t)i * Hn * DIn, h, nullptr, K, N, ntn, tiles);
    }
  }

  // final layernorm -> f32 (hfin)
  ln_kernel<<<(int)BL, Hn, 0, stream>>>(h, on_g, on_b, hfin, nullptr);

  attn_prep_kernel<<<Bn, Hn, 0, stream>>>(last, q_W, q_b, k_W, k_b, kq, qkb);
  attn_pool_kernel<<<Bn, 256, 0, stream>>>(hfin, x, kq, qkb, last, seq);
  head_kernel<<<Bn, 256, 0, stream>>>(seq, h1_W, h1_b, h2_W, h2_b, outp);

  (void)in_sizes; (void)n_in; (void)out_size; (void)ws_size;
}
